// InterSliceTextGuidedAttention_5952824672944
// MI455X (gfx1250) — matrix operand order repaired and checked by reading, not compiled, not run on hardware
//
#include <hip/hip_runtime.h>

typedef __attribute__((ext_vector_type(16))) __bf16 v16bf;
typedef __attribute__((ext_vector_type(8)))  float  v8f;

#define D_DIM 32
#define C_DIM 64
#define QKD   32
#define H_DIM 96
#define W_DIM 96
#define TW    8
#define BLOCK 256
#define NW    8        // waves per block (wave32)

__device__ __forceinline__ unsigned short f2bf(float f){
  union { __bf16 h; unsigned short u; } cv;
  cv.h = (__bf16)f;                       // native v_cvt_*_bf16_f32 on gfx1250
  return cv.u;
}
__device__ __forceinline__ float bf2f(unsigned short s){
  return __uint_as_float(((unsigned int)s) << 16);
}
// A-fragment (16-bit, 16x32) element e -> K offset, contiguous order k = 16*hi + e (matches B)
__device__ __forceinline__ int kofs(int e, int hi){
  return (hi << 4) | e;
}

union FA { v16bf v; unsigned short u[16]; };
union FB { v16bf v; uint4 q[2]; };

__device__ __forceinline__ v8f wmma_bf16(v16bf a, v16bf b, v8f c){
  return __builtin_amdgcn_wmma_f32_16x16x32_bf16(false, a, false, b, (short)0, c, false, false);
}

// A fragment from LDS matrix with row stride 32 (u16 units), rows = M, cols = K
__device__ __forceinline__ FA loadA_lds(const unsigned short* mat, int m0, int ln15, int hi){
  FA f; int m = m0 + ln15;
#pragma unroll
  for(int e = 0; e < 16; ++e) f.u[e] = mat[m * 32 + kofs(e, hi)];
  return f;
}
// A fragment (bf16) gathered from a global f32 weight matrix W[M][ldk]
__device__ __forceinline__ FA loadA_w(const float* W, int ldk, int m0, int k0, int ln15, int hi){
  FA f; int m = m0 + ln15;
#pragma unroll
  for(int e = 0; e < 16; ++e) f.u[e] = f2bf(W[m * ldk + k0 + kofs(e, hi)]);
  return f;
}
// B fragment: lane = column (N), 16 contiguous K values per half-wave -> two ds_load_b128
__device__ __forceinline__ v16bf loadB_lds(const unsigned short* mat, int col, int ld, int k0, int hi){
  FB f;
  const uint4* p = (const uint4*)(mat + col * ld + k0 + hi * 16);
  f.q[0] = p[0]; f.q[1] = p[1];
  return f.v;
}

// ---- kernel 0: text_bias[q] = bt[q] + sum_k wt[q,k] * mean_t(text[t,k]) ----
__global__ void text_bias_kernel(const float* __restrict__ text, const float* __restrict__ wt,
                                 const float* __restrict__ bt, float* __restrict__ tb){
  __shared__ float tm[256];
  int t = threadIdx.x;
  float s = 0.f;
  for(int r = 0; r < 32; ++r) s += text[r * 256 + t];
  tm[t] = s * (1.0f / 32.0f);
  __syncthreads();
  if(t < 32){
    float acc = bt[t];
    for(int k = 0; k < 256; ++k) acc += wt[t * 256 + k] * tm[k];
    tb[t] = acc;
  }
}

// ---- main fused attention kernel: one spatial location (b,h,w) per wave ----
__global__ __launch_bounds__(BLOCK) void attn_kernel(
    const float* __restrict__ x,   const float* __restrict__ wq, const float* __restrict__ bq,
    const float* __restrict__ wk,  const float* __restrict__ bk, const float* __restrict__ wvw,
    const float* __restrict__ bv,  const float* __restrict__ pos, const float* __restrict__ gamma,
    const float* __restrict__ tb,  float* __restrict__ out)
{
  __shared__ __align__(16) unsigned short xs [TW * D_DIM * C_DIM]; // 32KB: x bf16 [w][d][c]; later out stage [c][d][w]
  __shared__ __align__(16) unsigned short sQ [NW * D_DIM * QKD];   // 16KB per-wave: Q^T [d][q] -> attn^T [d][e]
  __shared__ __align__(16) unsigned short sKV[NW * D_DIM * QKD];   // 16KB per-wave: K [e][q] -> V m-tile [cl][e]

  const int tid  = threadIdx.x;
  const int lane = tid & 31;
  const int wvid = tid >> 5;       // wave 0..7 == location in tile
  const int ln15 = lane & 15;
  const int hi   = lane >> 4;

  const int blk = blockIdx.x;
  const int wt8 = blk % (W_DIM / TW);
  const int h   = (blk / (W_DIM / TW)) % H_DIM;
  const int b   = blk / ((W_DIM / TW) * H_DIM);
  const int w0  = wt8 * TW;

  const long dStride = (long)H_DIM * W_DIM;     // 9216
  const long cStride = (long)D_DIM * dStride;   // 294912
  const long bStride = (long)C_DIM * cStride;
  const long base    = (long)b * bStride + (long)h * W_DIM + w0;

  // ---- stage x tile into LDS as bf16 [w][d][c] ----
#pragma unroll 4
  for(int i = 0; i < (TW * D_DIM * C_DIM) / BLOCK; ++i){
    int flat = tid + BLOCK * i;
    int wl = flat & (TW - 1);
    int d  = (flat >> 3) & (D_DIM - 1);
    int c  = flat >> 8;
    float v = x[base + (long)c * cStride + (long)d * dStride + wl];
    xs[(wl * D_DIM + d) * C_DIM + c] = f2bf(v);
  }
  __syncthreads();

  const unsigned short* xloc = xs + wvid * (D_DIM * C_DIM);
  unsigned short* myQ  = sQ  + wvid * (D_DIM * QKD);
  unsigned short* myKV = sKV + wvid * (D_DIM * QKD);
  const v8f vzero = {0.f,0.f,0.f,0.f,0.f,0.f,0.f,0.f};
  const float RS = 0.17677669529663687f;   // 1/sqrt(QK)

  // ---- Q projection: D-frag (M=q, N=d) -> store Q^T to myQ [d][q] ----
#pragma unroll
  for(int mt = 0; mt < 2; ++mt){
    FA a0 = loadA_w(wq, C_DIM, mt * 16, 0,  ln15, hi);
    FA a1 = loadA_w(wq, C_DIM, mt * 16, 32, ln15, hi);
#pragma unroll
    for(int nt = 0; nt < 2; ++nt){
      v8f acc = wmma_bf16(a0.v, loadB_lds(xloc, nt * 16 + ln15, C_DIM, 0,  hi), vzero);
      acc     = wmma_bf16(a1.v, loadB_lds(xloc, nt * 16 + ln15, C_DIM, 32, hi), acc);
      int d = nt * 16 + ln15;
#pragma unroll
      for(int r = 0; r < 8; ++r){
        int q = mt * 16 + r + 8 * hi;
        myQ[d * QKD + q] = f2bf(acc[r] + bq[q] + pos[q * D_DIM + d] + tb[q]);
      }
    }
  }
  // ---- K projection -> myKV [e][q] ----
#pragma unroll
  for(int mt = 0; mt < 2; ++mt){
    FA a0 = loadA_w(wk, C_DIM, mt * 16, 0,  ln15, hi);
    FA a1 = loadA_w(wk, C_DIM, mt * 16, 32, ln15, hi);
#pragma unroll
    for(int nt = 0; nt < 2; ++nt){
      v8f acc = wmma_bf16(a0.v, loadB_lds(xloc, nt * 16 + ln15, C_DIM, 0,  hi), vzero);
      acc     = wmma_bf16(a1.v, loadB_lds(xloc, nt * 16 + ln15, C_DIM, 32, hi), acc);
      int d = nt * 16 + ln15;
#pragma unroll
      for(int r = 0; r < 8; ++r){
        int q = mt * 16 + r + 8 * hi;
        myKV[d * QKD + q] = f2bf(acc[r] + bk[q] + pos[q * D_DIM + d] + tb[q]);
      }
    }
  }
  __builtin_amdgcn_wave_barrier();   // DS ops are in-order per wave; fence compiler motion

  // ---- scores = Q^T K  (A from myQ [d][q], B from myKV [e][q]) ----
  v8f S[2][2];
#pragma unroll
  for(int mt = 0; mt < 2; ++mt){
    FA aq = loadA_lds(myQ, mt * 16, ln15, hi);
#pragma unroll
    for(int nt = 0; nt < 2; ++nt)
      S[mt][nt] = wmma_bf16(aq.v, loadB_lds(myKV, nt * 16 + ln15, QKD, 0, hi), vzero);
  }
  __builtin_amdgcn_wave_barrier();

  // ---- softmax over e (row d = r + 8*hi + 16*mt, col e = ln15 + 16*nt) ----
#pragma unroll
  for(int mt = 0; mt < 2; ++mt){
#pragma unroll
    for(int r = 0; r < 8; ++r){
      float s0 = S[mt][0][r], s1 = S[mt][1][r];
      float m = fmaxf(s0, s1);
      m = fmaxf(m, __shfl_xor(m, 8, 16));
      m = fmaxf(m, __shfl_xor(m, 4, 16));
      m = fmaxf(m, __shfl_xor(m, 2, 16));
      m = fmaxf(m, __shfl_xor(m, 1, 16));
      float e0 = __expf((s0 - m) * RS);
      float e1 = __expf((s1 - m) * RS);
      float sm = e0 + e1;
      sm += __shfl_xor(sm, 8, 16);
      sm += __shfl_xor(sm, 4, 16);
      sm += __shfl_xor(sm, 2, 16);
      sm += __shfl_xor(sm, 1, 16);
      float inv = 1.0f / sm;
      int d = mt * 16 + r + 8 * hi;
      myQ[d * D_DIM + ln15]      = f2bf(e0 * inv);   // attn^T staged as [d][e]
      myQ[d * D_DIM + 16 + ln15] = f2bf(e1 * inv);
    }
  }
  __builtin_amdgcn_wave_barrier();

  // ---- V projection + out GEMM, one 16-row c-tile at a time through myKV[0..511] ----
  v8f O[4][2];
#pragma unroll
  for(int mt = 0; mt < 4; ++mt){
    FA a0 = loadA_w(wvw, C_DIM, mt * 16, 0,  ln15, hi);
    FA a1 = loadA_w(wvw, C_DIM, mt * 16, 32, ln15, hi);
#pragma unroll
    for(int nt = 0; nt < 2; ++nt){
      v8f acc = wmma_bf16(a0.v, loadB_lds(xloc, nt * 16 + ln15, C_DIM, 0,  hi), vzero);
      acc     = wmma_bf16(a1.v, loadB_lds(xloc, nt * 16 + ln15, C_DIM, 32, hi), acc);
      int e = nt * 16 + ln15;
#pragma unroll
      for(int r = 0; r < 8; ++r){
        int cl = r + 8 * hi;                            // local row within c-tile
        myKV[cl * D_DIM + e] = f2bf(acc[r] + bv[mt * 16 + cl]);
      }
    }
    __builtin_amdgcn_wave_barrier();
    FA av = loadA_lds(myKV, 0, ln15, hi);               // V tile [cl][e]
#pragma unroll
    for(int nt = 0; nt < 2; ++nt)
      O[mt][nt] = wmma_bf16(av.v, loadB_lds(myQ, nt * 16 + ln15, D_DIM, 0, hi), vzero);
    __builtin_amdgcn_wave_barrier();                    // before next tile overwrites myKV
  }
  __syncthreads();                                      // all waves done reading xs

  // ---- stage O into xs as [c][d][w] for coalesced writeback ----
#pragma unroll
  for(int mt = 0; mt < 4; ++mt)
#pragma unroll
    for(int nt = 0; nt < 2; ++nt){
      int d = nt * 16 + ln15;
#pragma unroll
      for(int r = 0; r < 8; ++r){
        int c = mt * 16 + r + 8 * hi;
        xs[(c * D_DIM + d) * TW + wvid] = f2bf(O[mt][nt][r]);
      }
    }
  __syncthreads();

  // ---- gated residual, coalesced over w ----
  const float g = 1.0f / (1.0f + __expf(-gamma[0]));
#pragma unroll 4
  for(int i = 0; i < (TW * D_DIM * C_DIM) / BLOCK; ++i){
    int flat = tid + BLOCK * i;
    int wl = flat & (TW - 1);
    int d  = (flat >> 3) & (D_DIM - 1);
    int c  = flat >> 8;
    long ga = base + (long)c * cStride + (long)d * dStride + wl;
    float o = bf2f(xs[flat]);     // (c*32+d)*8+wl == flat
    out[ga] = g * o + (1.0f - g) * x[ga];
  }
}

extern "C" void kernel_launch(void* const* d_in, const int* in_sizes, int n_in,
                              void* d_out, int out_size, void* d_ws, size_t ws_size,
                              hipStream_t stream){
  const float* x    = (const float*)d_in[0];
  const float* text = (const float*)d_in[1];
  const float* wq   = (const float*)d_in[2];
  const float* bq   = (const float*)d_in[3];
  const float* wk   = (const float*)d_in[4];
  const float* bk   = (const float*)d_in[5];
  const float* wvw  = (const float*)d_in[6];
  const float* bv   = (const float*)d_in[7];
  const float* wt   = (const float*)d_in[8];
  const float* bt   = (const float*)d_in[9];
  const float* pos  = (const float*)d_in[10];
  const float* gam  = (const float*)d_in[11];
  float* out = (float*)d_out;
  float* tb  = (float*)d_ws;

  text_bias_kernel<<<1, 256, 0, stream>>>(text, wt, bt, tb);
  int blocks = 2 * H_DIM * (W_DIM / TW);    // 2304
  attn_kernel<<<blocks, BLOCK, 0, stream>>>(x, wq, bq, wk, bk, wvw, bv, pos, gam, tb, out);
}
